// PointNetSetAbstraction1_39951785787509
// MI455X (gfx1250) — matrix operand order repaired and checked by reading, not compiled, not run on hardware
//
#include <hip/hip_runtime.h>
#include <stdint.h>

// ---------------------------------------------------------------------------
// PointNet++ Set Abstraction for MI455X (gfx1250, wave32, WMMA)
// Activations stored column-major-per-sample [j][K] so WMMA B fragments are
// two contiguous b128 loads per lane (no LDS staging, no 2-byte gathers).
// ---------------------------------------------------------------------------
#define B_       16
#define N_       4096
#define NPOINT_  1024
#define NSAMPLE_ 32
#define R2_      0.04f
#define EPS_     1e-5f

static constexpr int S_ = B_ * NPOINT_ * NSAMPLE_;   // 524288 samples (columns)

typedef _Float16 v16h __attribute__((ext_vector_type(16)));
typedef _Float16 v8h  __attribute__((ext_vector_type(8)));
typedef float    v8f  __attribute__((ext_vector_type(8)));

union Frag16 { v16h v; v8h h[2]; };

// ds_swizzle_b32 butterfly add: xor-mask pattern, and_mask=0x1F (group of 32).
// Patterns 1,2,4,8 stay inside each 16-lane half (C-matrix row group).
#define SWZ_ADD(x, pat)                                                        \
  (x) += __int_as_float(__builtin_amdgcn_ds_swizzle(__float_as_int(x), (pat)))

// ---------------------------------------------------------------------------
// 0) prep: weights fp32 -> fp16 row-major [M][K] (layer0 K padded 6 -> 32),
//    zero the stats accumulators.
// ---------------------------------------------------------------------------
__global__ void prep_kernel(const float* __restrict__ w0,
                            const float* __restrict__ w1,
                            const float* __restrict__ w2,
                            _Float16* __restrict__ W0,
                            _Float16* __restrict__ W1,
                            _Float16* __restrict__ W2,
                            float* __restrict__ stats) {
  int t = blockIdx.x * 256 + threadIdx.x;
  if (t < 64 * 32) {               // W0: [64][32], channels 6..31 zero
    int r = t / 32, c = t % 32;
    W0[t] = (_Float16)((c < 6) ? w0[r * 6 + c] : 0.0f);
  }
  if (t < 64 * 64)  W1[t] = (_Float16)w1[t];      // [64][64]
  if (t < 128 * 64) W2[t] = (_Float16)w2[t];      // [128][64]
  if (t < 3 * 256)  stats[t] = 0.0f;              // 3 layers x (sum|sumsq)
}

// ---------------------------------------------------------------------------
// 1) Farthest point sampling: one workgroup per batch, dist[] in LDS,
//    1024 dependent iterations of update + block argmax.
// ---------------------------------------------------------------------------
__global__ void fps_kernel(const float* __restrict__ xyz,
                           float* __restrict__ newxyz,   // [B][NPOINT][3]
                           float* __restrict__ out) {    // d_out (B,3,NPOINT)
  const int b = blockIdx.x;
  const int t = threadIdx.x;                 // 256 threads
  const float* px = xyz + (size_t)b * 3 * N_;
  __shared__ float dist[N_];
  __shared__ float redV[256];
  __shared__ int   redI[256];
  __shared__ int   curIdx;
  __shared__ float cc[3];
  for (int i = t; i < N_; i += 256) dist[i] = 1e10f;
  if (t == 0) { curIdx = 0; cc[0] = px[0]; cc[1] = px[N_]; cc[2] = px[2 * N_]; }
  __syncthreads();
  for (int it = 0; it < NPOINT_; ++it) {
    const float cx = cc[0], cy = cc[1], cz = cc[2];
    if (t == 0) {
      float* nb = newxyz + ((size_t)b * NPOINT_ + it) * 3;
      nb[0] = cx; nb[1] = cy; nb[2] = cz;
      float* ob = out + (size_t)b * 3 * NPOINT_;
      ob[it] = cx; ob[NPOINT_ + it] = cy; ob[2 * NPOINT_ + it] = cz;
    }
    float bestV = -1.0f; int bestI = 0;
    for (int i = t; i < N_; i += 256) {
      float dx = px[i] - cx, dy = px[N_ + i] - cy, dz = px[2 * N_ + i] - cz;
      float d  = dx * dx + dy * dy + dz * dz;
      float dd = fminf(dist[i], d);
      dist[i]  = dd;
      if (dd > bestV) { bestV = dd; bestI = i; }
    }
    redV[t] = bestV; redI[t] = bestI;
    __syncthreads();
    for (int sdt = 128; sdt > 0; sdt >>= 1) {
      if (t < sdt) {
        float ov = redV[t + sdt]; int oi = redI[t + sdt];
        if (ov > redV[t] || (ov == redV[t] && oi < redI[t])) { redV[t] = ov; redI[t] = oi; }
      }
      __syncthreads();
    }
    if (t == 0) {
      int ni = redI[0];
      curIdx = ni;
      cc[0] = px[ni]; cc[1] = px[N_ + ni]; cc[2] = px[2 * N_ + ni];
    }
    __syncthreads();
  }
}

// ---------------------------------------------------------------------------
// 2) Ball query + grouping: one wave32 per centroid; ballot-scan the 4096
//    points in index order, keep first 32 with d2<=r2 (pad with first).
//    Emit X0 fp16 column-major [j][32] (channels 6..31 zero): each lane
//    writes its own 64B contiguous column (4 x b128).
// ---------------------------------------------------------------------------
__global__ void group_kernel(const float* __restrict__ xyz,
                             const float* __restrict__ pts,
                             const float* __restrict__ newxyz,
                             _Float16* __restrict__ X0) {
  const int w    = threadIdx.x >> 5;                 // wave in block (0..7)
  const int lane = threadIdx.x & 31;
  const int g    = blockIdx.x * 8 + w;               // centroid id
  const int b    = g / NPOINT_;
  __shared__ int sel[8][NSAMPLE_];

  const float cx = newxyz[(size_t)g * 3 + 0];
  const float cy = newxyz[(size_t)g * 3 + 1];
  const float cz = newxyz[(size_t)g * 3 + 2];
  const float* px = xyz + (size_t)b * 3 * N_;
  const float* pp = pts + (size_t)b * 3 * N_;

  int cnt = 0;
  for (int base = 0; base < N_ && cnt < NSAMPLE_; base += 32) {
    const int n = base + lane;
    float dx = px[n] - cx, dy = px[N_ + n] - cy, dz = px[2 * N_ + n] - cz;
    bool cond = (dx * dx + dy * dy + dz * dz) <= R2_;
    unsigned mm = __builtin_amdgcn_ballot_w32(cond);   // wave32 ballot
    int pos = cnt + __popc(mm & ((1u << lane) - 1u));
    if (cond && pos < NSAMPLE_) sel[w][pos] = n;
    cnt += __popc(mm);
  }
  if (cnt > NSAMPLE_) cnt = NSAMPLE_;
  asm volatile("s_wait_dscnt 0" ::: "memory");         // intra-wave LDS RAW
  const int first = sel[w][0];                          // centroid always hits
  const int myIdx = (lane < cnt) ? sel[w][lane] : first;

  _Float16 vals[32];
  vals[0] = (_Float16)(px[myIdx] - cx);
  vals[1] = (_Float16)(px[N_ + myIdx] - cy);
  vals[2] = (_Float16)(px[2 * N_ + myIdx] - cz);
  vals[3] = (_Float16)pp[myIdx];
  vals[4] = (_Float16)pp[N_ + myIdx];
  vals[5] = (_Float16)pp[2 * N_ + myIdx];
#pragma unroll
  for (int c = 6; c < 32; ++c) vals[c] = (_Float16)0.0f;

  // column j = g*32 + lane ; 32 contiguous halfs per column
  v8h* dst = (v8h*)(X0 + ((size_t)g * NSAMPLE_ + lane) * 32);
#pragma unroll
  for (int q = 0; q < 4; ++q) {
    v8h o;
#pragma unroll
    for (int e = 0; e < 8; ++e) o[e] = vals[q * 8 + e];
    dst[q] = o;
  }
}

// ---------------------------------------------------------------------------
// 3) WMMA GEMM  Y[M][S_] (fp16, row-major) = W[M][K] * X[K][S_],
//    X given column-major [j][K]. No LDS: A and B fragments are contiguous
//    b128 pairs straight from global (W in L0; X columns = 64/128B lines
//    reused by the 4 M-waves of each block). Per-channel sum/sumsq for
//    training-mode BN: nt-accumulate in registers, then ds_swizzle butterfly.
// ---------------------------------------------------------------------------
template <int KDIM>
__global__ void gemm_bn_kernel(const _Float16* __restrict__ Wm,
                               const _Float16* __restrict__ X,   // [S_][KDIM]
                               _Float16* __restrict__ Y,
                               float* __restrict__ stats,        // [sum M][sumsq M]
                               int Mtotal) {
  const int    t     = threadIdx.x;
  const int    mBase = blockIdx.y * 64;
  const size_t nBase = (size_t)blockIdx.x * 128;
  const int    w     = t >> 5, lane = t & 31;
  const int    wm    = (w & 3) * 16;      // wave M offset in tile
  const int    wn    = (w >> 2) * 64;     // wave N offset in tile
  const int    r     = lane & 15;
  const int    hs    = lane >> 4;         // lane half (K split)
  const int    row   = mBase + wm + r;

  v8f acc[4] = {};
#pragma unroll
  for (int k0 = 0; k0 < KDIM; k0 += 32) {
    // A fragment 16x32: lane row = M, two contiguous 8-half chunks
    Frag16 a;
    const _Float16* ap = Wm + (size_t)row * KDIM + k0;
    a.h[0] = *(const v8h*)(ap + hs * 8);
    a.h[1] = *(const v8h*)(ap + 16 + hs * 8);
#pragma unroll
    for (int nt = 0; nt < 4; ++nt) {
      // B fragment 32x16: lane column, two contiguous 8-half chunks in K
      const size_t col = nBase + wn + nt * 16 + r;
      const _Float16* bp = X + col * KDIM + k0 + hs * 8;
      Frag16 bf;
      bf.h[0] = *(const v8h*)(bp);
      bf.h[1] = *(const v8h*)(bp + 16);
      acc[nt] = __builtin_amdgcn_wmma_f32_16x16x32_f16(
          false, a.v, false, bf.v, (short)0, acc[nt], false, false);
    }
  }

  // Store fp16 Y (row-major); lanes of a row are contiguous columns.
#pragma unroll
  for (int nt = 0; nt < 4; ++nt) {
#pragma unroll
    for (int i = 0; i < 8; ++i) {
      const int m = wm + hs * 8 + i;     // C layout: lane<16 -> M=i, else M=8+i
      const int n = wn + nt * 16 + r;    // N = lane%16
      Y[(size_t)(mBase + m) * S_ + nBase + n] = (_Float16)acc[nt][i];
    }
  }

  // Per-channel stats: acc[nt][i] for all nt share row m, so fold nt in
  // registers first, then 4 ds_swizzle xor-butterfly steps over the 16 lanes.
#pragma unroll
  for (int i = 0; i < 8; ++i) {
    const int m = wm + hs * 8 + i;
    float s1 = acc[0][i] + acc[1][i] + acc[2][i] + acc[3][i];
    float s2 = acc[0][i] * acc[0][i] + acc[1][i] * acc[1][i] +
               acc[2][i] * acc[2][i] + acc[3][i] * acc[3][i];
    SWZ_ADD(s1, 0x041F); SWZ_ADD(s2, 0x041F);   // xor 1
    SWZ_ADD(s1, 0x081F); SWZ_ADD(s2, 0x081F);   // xor 2
    SWZ_ADD(s1, 0x101F); SWZ_ADD(s2, 0x101F);   // xor 4
    SWZ_ADD(s1, 0x201F); SWZ_ADD(s2, 0x201F);   // xor 8
    if (r == 0) {
      atomicAdd(&stats[mBase + m], s1);
      atomicAdd(&stats[Mtotal + mBase + m], s2);
    }
  }
}

// ---------------------------------------------------------------------------
// 4a) Per-channel BN coefficients from accumulated stats (bias cancels in BN).
// ---------------------------------------------------------------------------
__global__ void bn_coef_kernel(const float* __restrict__ stats,
                               const float* __restrict__ g,
                               const float* __restrict__ be,
                               float* __restrict__ coef, int M) {
  int c = blockIdx.x * 128 + threadIdx.x;
  if (c >= M) return;
  const float invS = 1.0f / (float)S_;
  const float mu   = stats[c] * invS;
  const float var  = stats[M + c] * invS - mu * mu;
  const float sc   = g[c] * rsqrtf(var + EPS_);
  coef[2 * c]     = sc;
  coef[2 * c + 1] = be[c] - mu * sc;
}

// ---------------------------------------------------------------------------
// 4b) BN + ReLU, transposing to the next layer's column-major input:
//     one thread per sample column j; reads Y[c][j] (coalesced across lanes),
//     writes 64 contiguous halfs (8 x b128) at Xn[j*64].
// ---------------------------------------------------------------------------
__global__ void bn_apply_kernel(const _Float16* __restrict__ Y,
                                _Float16* __restrict__ Xn,
                                const float* __restrict__ coef) {
  const int j = blockIdx.x * 256 + threadIdx.x;       // column (j < S_)
  v8h* dst = (v8h*)(Xn + (size_t)j * 64);
#pragma unroll
  for (int q = 0; q < 8; ++q) {
    v8h o;
#pragma unroll
    for (int e = 0; e < 8; ++e) {
      const int c = q * 8 + e;
      float z = (float)Y[(size_t)c * S_ + j] * coef[2 * c] + coef[2 * c + 1];
      o[e] = (_Float16)fmaxf(z, 0.0f);
    }
    dst[q] = o;
  }
}

// ---------------------------------------------------------------------------
// 5) Final BN + ReLU + max over the 32 neighbors -> new_feat (B,128,NPOINT).
// ---------------------------------------------------------------------------
__global__ void bn_max_kernel(const _Float16* __restrict__ Y2,
                              float* __restrict__ out_feat,
                              const float* __restrict__ coef) {
  const int tid = blockIdx.x * 256 + threadIdx.x;    // b*128*1024 + ch*1024 + s
  if (tid >= B_ * 128 * NPOINT_) return;
  const int s  = tid % NPOINT_;
  const int ch = (tid / NPOINT_) % 128;
  const int b  = tid / (NPOINT_ * 128);
  const float sc = coef[2 * ch], sh = coef[2 * ch + 1];
  const size_t jb = ((size_t)b * NPOINT_ + s) * NSAMPLE_;
  const v8h* p = (const v8h*)(Y2 + (size_t)ch * S_ + jb);
  float mx = 0.0f;   // ReLU floor: max(relu(z)) == relu(max(z))
#pragma unroll
  for (int q = 0; q < 4; ++q) {
    v8h v = p[q];
#pragma unroll
    for (int e = 0; e < 8; ++e) mx = fmaxf(mx, (float)v[e] * sc + sh);
  }
  out_feat[tid] = mx;
}

// ---------------------------------------------------------------------------
// Launch: prep -> FPS -> group -> [GEMM -> coef -> BN]x2 -> GEMM -> coef -> max
// ---------------------------------------------------------------------------
extern "C" void kernel_launch(void* const* d_in, const int* in_sizes, int n_in,
                              void* d_out, int out_size, void* d_ws, size_t ws_size,
                              hipStream_t stream) {
  const float* xyz = (const float*)d_in[0];
  const float* pts = (const float*)d_in[1];
  const float* w0  = (const float*)d_in[2];
  const float* g0  = (const float*)d_in[4];
  const float* be0 = (const float*)d_in[5];
  const float* w1  = (const float*)d_in[6];
  const float* g1  = (const float*)d_in[8];
  const float* be1 = (const float*)d_in[9];
  const float* w2  = (const float*)d_in[10];
  const float* g2  = (const float*)d_in[12];
  const float* be2 = (const float*)d_in[13];
  float* out = (float*)d_out;

  char* ws = (char*)d_ws;
  _Float16* W0    = (_Float16*)(ws);                    //  4 KB
  _Float16* W1    = (_Float16*)(ws + 4096);             //  8 KB
  _Float16* W2    = (_Float16*)(ws + 12288);            // 16 KB
  float*    stats = (float*)(ws + 28672);               // 3x256 floats
  float*    coef  = (float*)(ws + 32768);               // 256 floats (reused)
  float*    nxyz  = (float*)(ws + 36864);               // 192 KB
  _Float16* Xbuf  = (_Float16*)(ws + ((size_t)1 << 20));                      // <=67 MB
  _Float16* Ybuf  = (_Float16*)(ws + ((size_t)1 << 20) + ((size_t)68 << 20)); // <=134 MB

  prep_kernel<<<32, 256, 0, stream>>>(w0, w1, w2, W0, W1, W2, stats);
  fps_kernel<<<B_, 256, 0, stream>>>(xyz, nxyz, out);
  group_kernel<<<(B_ * NPOINT_) / 8, 256, 0, stream>>>(xyz, pts, nxyz, Xbuf);

  const dim3 gemmGrid1(S_ / 128, 1), gemmGrid2(S_ / 128, 2);
  // layer 0: K=32 (padded from 6), M=64
  gemm_bn_kernel<32><<<gemmGrid1, 256, 0, stream>>>(W0, Xbuf, Ybuf, stats, 64);
  bn_coef_kernel<<<1, 128, 0, stream>>>(stats, g0, be0, coef, 64);
  bn_apply_kernel<<<S_ / 256, 256, 0, stream>>>(Ybuf, Xbuf, coef);
  // layer 1: K=64, M=64
  gemm_bn_kernel<64><<<gemmGrid1, 256, 0, stream>>>(W1, Xbuf, Ybuf, stats + 256, 64);
  bn_coef_kernel<<<1, 128, 0, stream>>>(stats + 256, g1, be1, coef, 64);
  bn_apply_kernel<<<S_ / 256, 256, 0, stream>>>(Ybuf, Xbuf, coef);
  // layer 2: K=64, M=128
  gemm_bn_kernel<64><<<gemmGrid2, 256, 0, stream>>>(W2, Xbuf, Ybuf, stats + 512, 128);
  bn_coef_kernel<<<1, 128, 0, stream>>>(stats + 512, g2, be2, coef, 128);
  bn_max_kernel<<<(B_ * 128 * NPOINT_) / 256, 256, 0, stream>>>(
      Ybuf, out + (size_t)B_ * 3 * NPOINT_, coef);
}
